// SpatioTemporalAttention_56246891708482
// MI455X (gfx1250) — matrix operand order repaired and checked by reading, not compiled, not run on hardware
//
#include <hip/hip_runtime.h>
#include <hip/hip_bf16.h>
#include <stdint.h>

// ---------------- problem constants (from reference) ----------------
#define NB   8          // batch
#define FF   16         // frames
#define SS   196        // spatial
#define DD   1024       // model dim
#define HH   8          // heads
#define HDIM 128        // head dim
#define LTOK (1 + FF * SS)      // 3137 tokens
#define MTOT (NB * LTOK)        // 25096 rows

// ---------------- WMMA types ----------------
typedef __attribute__((ext_vector_type(16))) __bf16 v16bf;
typedef __attribute__((ext_vector_type(8)))  float  v8f;

union Frag { unsigned int u[8]; v16bf v; };

__device__ __forceinline__ unsigned short f32_to_bf16_rne(float f) {
    unsigned int u = __float_as_uint(f);
    unsigned int lsb = (u >> 16) & 1u;
    u += 0x7fffu + lsb;                 // round to nearest even
    return (unsigned short)(u >> 16);
}

// ---------------- f32 -> bf16 bulk conversion (n multiple of 4) ----------------
__global__ __launch_bounds__(256) void cvt_f32_bf16(
    const float* __restrict__ in, unsigned short* __restrict__ out, long n)
{
    long i = ((long)blockIdx.x * 256 + threadIdx.x) * 4;
    if (i >= n) return;
    float4 f = *(const float4*)(in + i);
    ushort4 o;
    o.x = f32_to_bf16_rne(f.x);
    o.y = f32_to_bf16_rne(f.y);
    o.z = f32_to_bf16_rne(f.z);
    o.w = f32_to_bf16_rne(f.w);
    *(ushort4*)(out + i) = o;
}

// ---------------- GEMM: C[M,N](f32) = A[M,K](bf16) @ B[K,N](bf16) + bias ----------------
// 128x128 block, BK=64. 8 waves in 4(M) x 2(N) grid; each wave: 32x64 = 2x4 WMMA
// tiles, 2 K-substeps per stage => 16 v_wmma per wave per barrier.
#define BM 128
#define BN 128
#define BK 64
#define LDSP 66   // padded K-stride in shorts (4B-aligned pairs, de-banked rows)

__global__ __launch_bounds__(256) void gemm_bf16_wmma(
    const unsigned short* __restrict__ A, const unsigned short* __restrict__ B,
    const float* __restrict__ bias, float* __restrict__ C,
    int M, int N, int K)
{
    __shared__ __align__(16) unsigned short As[BM][LDSP];    // [m][k]
    __shared__ __align__(16) unsigned short BsT[BN][LDSP];   // [n][k] (transposed)

    const int tid    = threadIdx.x;
    const int lane   = tid & 31;
    const int wave   = tid >> 5;          // 0..7
    const int laneHi = (lane >> 4) & 1;   // 0/1
    const int lane15 = lane & 15;

    const int wm = wave >> 1;             // 0..3  -> 32-row strip
    const int wn = wave & 1;              // 0..1  -> 64-col strip

    const int block_m = blockIdx.x * BM;
    const int block_n = blockIdx.y * BN;

    // ---- loop-invariant staging indices ----
    // A: 128 rows, 2 threads per row, 32 bf16 each
    const int ar   = tid >> 1;
    const int acol = (tid & 1) * 32;
    int arow = block_m + ar;
    if (arow >= M) arow = M - 1;          // clamp (stores are guarded; no divergence)
    const unsigned short* aptr = A + (size_t)arow * K + acol;

    // B: 64 k-rows, 4 threads per row, 32 bf16 each
    const int br   = tid >> 2;
    const int bcol = (tid & 3) * 32;
    const unsigned short* bptr = B + (size_t)br * N + block_n + bcol;

    v8f acc[2][4] = {};

    for (int k0 = 0; k0 < K; k0 += BK) {
        // ---- prefetch next tiles (global_prefetch_b8) ----
        if (k0 + BK < K) {
            __builtin_prefetch((const void*)(aptr + k0 + BK), 0, 1);
            __builtin_prefetch((const void*)(bptr + (size_t)(k0 + BK) * N), 0, 1);
        }
        // ---- stage A tile (128 x 64 bf16): 4 b128 loads/thread ----
        {
            union { uint4 q[4]; unsigned int u[16]; } ua;
            const unsigned short* ap = aptr + k0;
            ua.q[0] = *(const uint4*)(ap);
            ua.q[1] = *(const uint4*)(ap + 8);
            ua.q[2] = *(const uint4*)(ap + 16);
            ua.q[3] = *(const uint4*)(ap + 24);
            #pragma unroll
            for (int p = 0; p < 16; ++p)
                *(unsigned int*)&As[ar][acol + 2 * p] = ua.u[p];
        }
        // ---- stage B tile (64 x 128 bf16) transposed into BsT[n][k] ----
        {
            union { uint4 q[4]; unsigned short s[32]; } ub;
            const unsigned short* bp = bptr + (size_t)k0 * N;
            ub.q[0] = *(const uint4*)(bp);
            ub.q[1] = *(const uint4*)(bp + 8);
            ub.q[2] = *(const uint4*)(bp + 16);
            ub.q[3] = *(const uint4*)(bp + 24);
            #pragma unroll
            for (int i = 0; i < 32; ++i)
                BsT[bcol + i][br] = ub.s[i];
        }
        __syncthreads();

        // ---- two K=32 substeps per stage ----
        #pragma unroll
        for (int ks = 0; ks < BK; ks += 32) {
            // A fragments (CDNA5 16-bit layout):
            // lane<16: row=lane15, K in {0..7,16..23}; lane>=16: K+8
            Frag afrag[2], bfrag[4];
            #pragma unroll
            for (int i = 0; i < 2; ++i) {
                const unsigned short* arp = &As[wm * 32 + i * 16 + lane15][ks];
                #pragma unroll
                for (int p = 0; p < 8; ++p) {
                    int kk = ((p < 4) ? 2 * p : 2 * p + 8) + (laneHi ? 8 : 0);
                    afrag[i].u[p] = *(const unsigned int*)(arp + kk);
                }
            }
            // B fragments: lane<16 -> N=lane15, K in {0..7,16..23}; lane>=16: K+8
            #pragma unroll
            for (int j = 0; j < 4; ++j) {
                const unsigned short* brp = &BsT[wn * 64 + j * 16 + lane15][ks];
                #pragma unroll
                for (int p = 0; p < 8; ++p) {
                    int kk = ((p < 4) ? 2 * p : 2 * p + 8) + (laneHi ? 8 : 0);
                    bfrag[j].u[p] = *(const unsigned int*)(brp + kk);
                }
            }
            // 2x4 WMMA tiles
            #pragma unroll
            for (int i = 0; i < 2; ++i)
                #pragma unroll
                for (int j = 0; j < 4; ++j)
                    acc[i][j] = __builtin_amdgcn_wmma_f32_16x16x32_bf16(
                        false, afrag[i].v, false, bfrag[j].v,
                        (short)0, acc[i][j], false, false);
        }
        __syncthreads();
    }

    // ---- epilogue: C/D layout (VGPR r -> M = r + 8*laneHi; lane15 -> N) ----
    #pragma unroll
    for (int i = 0; i < 2; ++i) {
        #pragma unroll
        for (int j = 0; j < 4; ++j) {
            int col = block_n + wn * 64 + j * 16 + lane15;
            float bv = bias ? bias[col] : 0.0f;
            #pragma unroll
            for (int r = 0; r < 8; ++r) {
                int row = block_m + wm * 32 + i * 16 + r + laneHi * 8;
                if (row < M)
                    C[(size_t)row * N + col] = acc[i][j][r] + bv;
            }
        }
    }
}

// ---------------- in-place LayerNorm over last dim (D=1024) ----------------
__global__ __launch_bounds__(256) void layernorm_inplace(
    float* __restrict__ X, const float* __restrict__ g, const float* __restrict__ b)
{
    __shared__ float rs[256], rss[256];
    const int tid = threadIdx.x;
    float* x = X + (size_t)blockIdx.x * DD;

    float vals[4];
    float s = 0.f, ss = 0.f;
    #pragma unroll
    for (int i = 0; i < 4; ++i) {
        float f = x[tid + i * 256];
        vals[i] = f; s += f; ss += f * f;
    }
    rs[tid] = s; rss[tid] = ss;
    __syncthreads();
    for (int off = 128; off > 0; off >>= 1) {
        if (tid < off) { rs[tid] += rs[tid + off]; rss[tid] += rss[tid + off]; }
        __syncthreads();
    }
    float mean = rs[0] * (1.0f / DD);
    float var  = rss[0] * (1.0f / DD) - mean * mean;
    float inv  = rsqrtf(var + 1e-5f);
    #pragma unroll
    for (int i = 0; i < 4; ++i) {
        int c = tid + i * 256;
        x[c] = (vals[i] - mean) * inv * g[c] + b[c];
    }
}

// ---------------- temporal attention (att_type == 1) ----------------
// One block per (head-batch, spatial pos): q 16x128 vs [feat; 16 frames] keys.
__global__ __launch_bounds__(256) void attn_temporal(
    const float* __restrict__ Q, const float* __restrict__ Km,
    const float* __restrict__ Vm, float* __restrict__ O)
{
    __shared__ float qs[FF][HDIM];
    __shared__ float ks[FF + 1][HDIM];
    __shared__ float vs[FF + 1][HDIM];
    __shared__ float ps[FF][FF + 1];

    const int s  = blockIdx.x;       // 0..195
    const int bh = blockIdx.y;       // 0..63
    const int b  = bh >> 3, h = bh & 7;
    const int tid = threadIdx.x;

    const size_t base = (size_t)b * LTOK * DD + (size_t)h * HDIM;

    for (int idx = tid; idx < FF * HDIM; idx += 256) {
        int f = idx >> 7, d = idx & 127;
        size_t l = 1 + (size_t)f * SS + s;
        qs[f][d] = Q[base + l * DD + d];
    }
    for (int idx = tid; idx < (FF + 1) * HDIM; idx += 256) {
        int j = idx >> 7, d = idx & 127;
        size_t l = (j == 0) ? 0 : (1 + (size_t)(j - 1) * SS + s);
        ks[j][d] = Km[base + l * DD + d];
        vs[j][d] = Vm[base + l * DD + d];
    }
    __syncthreads();

    const float scale = 0.088388347648318447f;   // 1/sqrt(128)
    for (int idx = tid; idx < FF * (FF + 1); idx += 256) {
        int f = idx / (FF + 1), j = idx % (FF + 1);
        float acc = 0.f;
        #pragma unroll 8
        for (int d = 0; d < HDIM; ++d) acc += qs[f][d] * ks[j][d];
        ps[f][j] = acc * scale;
    }
    __syncthreads();

    if (tid < FF) {
        float mx = ps[tid][0];
        #pragma unroll
        for (int j = 1; j <= FF; ++j) mx = fmaxf(mx, ps[tid][j]);
        float sum = 0.f;
        #pragma unroll
        for (int j = 0; j <= FF; ++j) { float e = __expf(ps[tid][j] - mx); ps[tid][j] = e; sum += e; }
        float r = 1.0f / sum;
        #pragma unroll
        for (int j = 0; j <= FF; ++j) ps[tid][j] *= r;
    }
    __syncthreads();

    for (int idx = tid; idx < FF * HDIM; idx += 256) {
        int f = idx >> 7, d = idx & 127;
        float acc = 0.f;
        #pragma unroll
        for (int j = 0; j <= FF; ++j) acc += ps[f][j] * vs[j][d];
        size_t l = 1 + (size_t)f * SS + s;
        O[base + l * DD + d] = acc;
    }
}

// ---------------- splice feature-query token through (l == 0 rows) ----------------
__global__ __launch_bounds__(256) void copy_feat(
    const float* __restrict__ Q, float* __restrict__ O)
{
    int idx = blockIdx.x * 256 + threadIdx.x;  // 8*1024 total
    int b = idx >> 10, d = idx & 1023;
    size_t off = (size_t)b * LTOK * DD + d;
    O[off] = Q[off];
}

// ---------------- launcher ----------------
extern "C" void kernel_launch(void* const* d_in, const int* in_sizes, int n_in,
                              void* d_out, int out_size, void* d_ws, size_t ws_size,
                              hipStream_t stream) {
    const float* x       = (const float*)d_in[0];
    const float* W_in    = (const float*)d_in[1];
    const float* b_in    = (const float*)d_in[2];
    const float* g_in    = (const float*)d_in[3];
    const float* beta_in = (const float*)d_in[4];
    const float* W_q     = (const float*)d_in[5];
    const float* W_k     = (const float*)d_in[6];
    const float* W_v     = (const float*)d_in[7];
    const float* g_out   = (const float*)d_in[8];
    const float* beta_out= (const float*)d_in[9];
    const float* W_out   = (const float*)d_in[10];
    const float* b_out   = (const float*)d_in[11];
    (void)in_sizes; (void)n_in; (void)out_size; (void)ws_size;

    const int M = MTOT;                  // 25096
    const size_t CNT = (size_t)M * DD;   // floats per activation buffer
    const size_t WCNT = (size_t)DD * DD; // weight elements

    // f32 workspace buffers
    float* x0 = (float*)d_ws;            // stage0 / attention output (reused)
    float* q  = x0 + CNT;
    float* k  = x0 + 2 * CNT;
    float* v  = x0 + 3 * CNT;
    // bf16 workspace buffers
    unsigned short* xb  = (unsigned short*)(x0 + 4 * CNT);  // activations (reused)
    unsigned short* wIn = xb + CNT;
    unsigned short* wQ  = wIn + WCNT;
    unsigned short* wK  = wQ + WCNT;
    unsigned short* wV  = wK + WCNT;
    unsigned short* wO  = wV + WCNT;

    dim3 gblk((M + BM - 1) / BM, DD / BN);   // 197 x 8
    dim3 tblk(256);
    const int cact = (int)(CNT / 4 / 256);
    const int cwt  = (int)(WCNT / 4 / 256);

    // pre-convert all five weights to bf16 once
    cvt_f32_bf16<<<cwt, tblk, 0, stream>>>(W_in,  wIn, (long)WCNT);
    cvt_f32_bf16<<<cwt, tblk, 0, stream>>>(W_q,   wQ,  (long)WCNT);
    cvt_f32_bf16<<<cwt, tblk, 0, stream>>>(W_k,   wK,  (long)WCNT);
    cvt_f32_bf16<<<cwt, tblk, 0, stream>>>(W_v,   wV,  (long)WCNT);
    cvt_f32_bf16<<<cwt, tblk, 0, stream>>>(W_out, wO,  (long)WCNT);

    // x0 = LN(x @ W_in + b_in)
    cvt_f32_bf16<<<cact, tblk, 0, stream>>>(x, xb, (long)CNT);
    gemm_bf16_wmma<<<gblk, tblk, 0, stream>>>(xb, wIn, b_in, x0, M, DD, DD);
    layernorm_inplace<<<M, tblk, 0, stream>>>(x0, g_in, beta_in);

    // projections (single bf16 conversion reused 3x)
    cvt_f32_bf16<<<cact, tblk, 0, stream>>>(x0, xb, (long)CNT);
    gemm_bf16_wmma<<<gblk, tblk, 0, stream>>>(xb, wQ, nullptr, q, M, DD, DD);
    gemm_bf16_wmma<<<gblk, tblk, 0, stream>>>(xb, wK, nullptr, k, M, DD, DD);
    gemm_bf16_wmma<<<gblk, tblk, 0, stream>>>(xb, wV, nullptr, v, M, DD, DD);

    // temporal attention; output overwrites x0 (fully covered: l=0 via copy_feat,
    // l in [1, L) via attn blocks)
    attn_temporal<<<dim3(SS, NB * HH), tblk, 0, stream>>>(q, k, v, x0);
    copy_feat<<<(NB * DD) / 256, tblk, 0, stream>>>(q, x0);

    // out = LN(x0) @ W_out + b_out
    layernorm_inplace<<<M, tblk, 0, stream>>>(x0, g_out, beta_out);
    cvt_f32_bf16<<<cact, tblk, 0, stream>>>(x0, xb, (long)CNT);
    gemm_bf16_wmma<<<gblk, tblk, 0, stream>>>(xb, wO, b_out, (float*)d_out, M, DD, DD);
}
